// Non_Local_Block_72464688218298
// MI455X (gfx1250) — matrix operand order repaired and checked by reading, not compiled, not run on hardware
//
#include <hip/hip_runtime.h>
#include <hip/hip_bf16.h>
#include <stdint.h>

typedef __attribute__((ext_vector_type(16))) _Float16 v16h;
typedef __attribute__((ext_vector_type(8)))  _Float16 v8h;
typedef __attribute__((ext_vector_type(8)))  float    v8f;
typedef __attribute__((ext_vector_type(4)))  unsigned int v4u;
typedef __attribute__((ext_vector_type(4)))  int v4i;

constexpr int kD    = 256;      // channels
constexpr int kN    = 65536;    // points per batch
constexpr int kB    = 2;        // batch
constexpr int kPOS  = kB * kN;  // 131072 total positions

// ---------------- optional CDNA5 async global->LDS path (probe via __has_builtin) -----------
#if defined(__has_builtin)
#  if __has_builtin(__builtin_amdgcn_global_load_async_to_lds_b128) && \
      __has_builtin(__builtin_amdgcn_s_wait_asynccnt)
#    define USE_ASYNC_LDS 1
#  endif
#endif

__device__ __forceinline__ void cp16(const _Float16* g, _Float16* l) {
#ifdef USE_ASYNC_LDS
  __builtin_amdgcn_global_load_async_to_lds_b128(
      (__attribute__((address_space(1))) v4i*)(void*)g,
      (__attribute__((address_space(3))) v4i*)(void*)l, 0, 0);
#else
  *(v4u*)l = *(const v4u*)g;
#endif
}

__device__ __forceinline__ void cp_wait() {
#ifdef USE_ASYNC_LDS
  __builtin_amdgcn_s_wait_asynccnt(0);
#endif
}

// ---------------- WMMA fragment helpers (CDNA5 16x16x32 f16, wave32) ------------------------
union HF16 { v16h v; v8h h[2]; };

// A-matrix 16x32 f16: lane&15 = M row; lanes>=16 take K+8; VGPR0-3 = K 0..7, VGPR4-7 = K 16..23
__device__ __forceinline__ v16h load_afrag(const _Float16* base, int rbase, int stride,
                                           int k0, int lane) {
  int r = rbase + (lane & 15);
  int g = (lane >> 4) << 3;             // 0 or 8
  const _Float16* p = base + (size_t)r * stride + k0 + g;
  HF16 u;
  u.h[0] = *(const v8h*)p;              // K = k0+g   .. +7
  u.h[1] = *(const v8h*)(p + 16);       // K = k0+g+16.. +7
  return u.v;
}

// B-matrix 32x16 f16: lane&15 = N col; lanes>=16 take K+8; VGPR0-3 = K 0..7, VGPR4-7 = K 16..23
__device__ __forceinline__ v16h load_bfrag(const _Float16* base, int nbase, int stride,
                                           int k0, int lane) {
  int n = nbase + (lane & 15);
  int g = (lane >> 4) << 3;             // 0 or 8
  const _Float16* p = base + (size_t)n * stride + k0 + g;
  HF16 u;
  u.h[0] = *(const v8h*)p;              // K = k0+g   .. +7
  u.h[1] = *(const v8h*)(p + 16);       // K = k0+g+16.. +7
  return u.v;
}

__device__ __forceinline__ v8f wmma_f16(v16h a, v16h b, v8f c) {
  return __builtin_amdgcn_wmma_f32_16x16x32_f16(false, a, false, b, (short)0, c, false, false);
}

// ---------------- kernel 1: fold BN into conv weights ---------------------------------------
__global__ void prep_kernel(const float* __restrict__ W, const float* __restrict__ bb,
                            const float* __restrict__ gmm, const float* __restrict__ bet,
                            const float* __restrict__ mn, const float* __restrict__ vr,
                            _Float16* __restrict__ Wp, float* __restrict__ bp) {
  int po = blockIdx.x;                  // [0, 6*256): (proj, out-channel)
  int t  = threadIdx.x;                 // [0, 256)
  float scale = gmm[po] * rsqrtf(vr[po] + 1e-5f);
  Wp[(size_t)po * kD + t] = (_Float16)(W[(size_t)po * kD + t] * scale);
  if (t == 0) bp[po] = (bb[po] - mn[po]) * scale + bet[po];
}

// ---------------- kernel 2: feature [B,d,N] f32 -> Xt [B*N, d] f16 --------------------------
__global__ void feat_t_kernel(const float* __restrict__ in, _Float16* __restrict__ out) {
  __shared__ float tile[32][33];
  int n0 = blockIdx.x * 32, d0 = blockIdx.y * 32, b = blockIdx.z;
  int tx = threadIdx.x, ty = threadIdx.y;
#pragma unroll
  for (int i = 0; i < 4; ++i) {
    int dd = d0 + ty + i * 8;
    tile[ty + i * 8][tx] = in[(size_t)(b * kD + dd) * kN + n0 + tx];
  }
  __syncthreads();
#pragma unroll
  for (int i = 0; i < 4; ++i) {
    int nn = ty + i * 8;
    out[(size_t)(b * kN + n0 + nn) * kD + d0 + tx] = (_Float16)tile[tx][nn];
  }
}

// ---------------- kernel 3: fused conv1x1+BN+ReLU GEMM  Y[p,o] = relu(X[p,:]·W'[o,:] + b') ---
// block: 256 thr (8 waves), 128 positions x 256 outputs, K=256. grid.y selects q/k/v.
__global__ __launch_bounds__(256, 1)
void proj_kernel(const _Float16* __restrict__ Xin, const _Float16* __restrict__ Wp,
                 const float* __restrict__ bp,
                 _Float16* __restrict__ Yq, _Float16* __restrict__ Yk,
                 _Float16* __restrict__ Yv, int transposeOut) {
  __shared__ __align__(16) _Float16 Wsh[64 * kD];            // 32 KB
  int tid = threadIdx.x, lane = tid & 31, wave = tid >> 5;
  int proj = blockIdx.y;
  const _Float16* Wm = Wp + (size_t)proj * kD * kD;
  const float*    bm = bp + proj * kD;
  _Float16* Y = proj == 0 ? Yq : (proj == 1 ? Yk : Yv);
  int posBase = blockIdx.x * 128;
  int myrow   = posBase + wave * 16;

  v16h xf[8];
#pragma unroll
  for (int dc = 0; dc < 8; ++dc) xf[dc] = load_afrag(Xin, myrow, kD, dc * 32, lane);

  v8f acc[16];
#pragma unroll
  for (int i = 0; i < 16; ++i) acc[i] = (v8f)0.0f;

  for (int oc = 0; oc < 4; ++oc) {                           // stream 64 W' rows at a time
#pragma unroll
    for (int i = 0; i < 8; ++i) {
      int idx = tid + i * 256;
      int row = idx >> 4, c16 = idx & 15;
      cp16(Wm + (size_t)(oc * 64 + row) * kD + c16 * 16, Wsh + row * kD + c16 * 16);
    }
    cp_wait();
    __syncthreads();
#pragma unroll
    for (int ot = 0; ot < 4; ++ot)
#pragma unroll
      for (int dc = 0; dc < 8; ++dc) {
        v16h wf = load_bfrag(Wsh, ot * 16, kD, dc * 32, lane);
        acc[oc * 4 + ot] = wmma_f16(xf[dc], wf, acc[oc * 4 + ot]);
      }
    __syncthreads();
  }

  int n = lane & 15, hh = lane >> 4;
#pragma unroll
  for (int t16 = 0; t16 < 16; ++t16) {
    int o = t16 * 16 + n;
    float bias = bm[o];
#pragma unroll
    for (int r = 0; r < 8; ++r) {
      int p = myrow + r + 8 * hh;
      float val = acc[t16][r] + bias;
      val = val > 0.f ? val : 0.f;
      int pout;
      if (transposeOut) {                                    // (g,r)->(r,g) within batch
        int b2 = p >> 16, pb = p & 65535;
        pout = (b2 << 16) | ((pb & 255) << 8) | (pb >> 8);
      } else pout = p;
      Y[(size_t)pout * kD + o] = (_Float16)val;
    }
  }
}

// ---------------- kernel 4: attention, one block per 128 rows of one 256x256 group ----------
__global__ __launch_bounds__(256, 1)
void attn_kernel(const _Float16* __restrict__ Qg, const _Float16* __restrict__ Kg,
                 const _Float16* __restrict__ Vg, _Float16* __restrict__ Og) {
  __shared__ __align__(16) _Float16 Kbuf[64 * kD];           // 32 KB, K chunk
  __shared__ __align__(16) _Float16 Pbuf[8 * 16 * kD];       // 64 KB, per-wave P strips
  __shared__ __align__(16) _Float16 Vt[kD * 32];             // 16 KB, V chunk transposed
  int tid = threadIdx.x, lane = tid & 31, wave = tid >> 5;
  int grp0  = blockIdx.x * 256;                              // first row of group
  int myrow = grp0 + blockIdx.y * 128 + wave * 16;

  v16h qf[8];
#pragma unroll
  for (int dc = 0; dc < 8; ++dc) qf[dc] = load_afrag(Qg, myrow, kD, dc * 32, lane);

  v8f S[16];
#pragma unroll
  for (int i = 0; i < 16; ++i) S[i] = (v8f)0.0f;

  // ---- pass 1: S = Q K^T over 4 chunks of 64 key-rows ----
  for (int kc = 0; kc < 4; ++kc) {
#pragma unroll
    for (int i = 0; i < 8; ++i) {
      int idx = tid + i * 256;
      int row = idx >> 4, c16 = idx & 15;
      cp16(Kg + (size_t)(grp0 + kc * 64 + row) * kD + c16 * 16, Kbuf + row * kD + c16 * 16);
    }
    cp_wait();
    __syncthreads();
#pragma unroll
    for (int tt = 0; tt < 4; ++tt)
#pragma unroll
      for (int dc = 0; dc < 8; ++dc) {
        v16h kf = load_bfrag(Kbuf, tt * 16, kD, dc * 32, lane);
        S[kc * 4 + tt] = wmma_f16(qf[dc], kf, S[kc * 4 + tt]);
      }
    __syncthreads();
  }

  // ---- rowwise softmax: component r = rows r (lanes 0-15) and r+8 (lanes 16-31) ----
#pragma unroll
  for (int r = 0; r < 8; ++r) {
    float mx = -3.0e38f;
#pragma unroll
    for (int ct = 0; ct < 16; ++ct) mx = fmaxf(mx, S[ct][r]);
#pragma unroll
    for (int off = 8; off >= 1; off >>= 1) mx = fmaxf(mx, __shfl_xor(mx, off, 32));
    float sum = 0.f;
#pragma unroll
    for (int ct = 0; ct < 16; ++ct) { float e = __expf(S[ct][r] - mx); S[ct][r] = e; sum += e; }
#pragma unroll
    for (int off = 8; off >= 1; off >>= 1) sum += __shfl_xor(sum, off, 32);
    float inv = 1.0f / sum;
#pragma unroll
    for (int ct = 0; ct < 16; ++ct) S[ct][r] *= inv;
  }

  // ---- P -> per-wave LDS strip (f16, row-major 16x256) for A-fragment reload ----
  _Float16* Prow = Pbuf + (size_t)wave * 16 * kD;
  {
    int n = lane & 15, hh = lane >> 4;
#pragma unroll
    for (int ct = 0; ct < 16; ++ct)
#pragma unroll
      for (int r = 0; r < 8; ++r)
        Prow[(r + 8 * hh) * kD + ct * 16 + n] = (_Float16)S[ct][r];
  }
  __syncthreads();

  // ---- pass 2: O = P V over 8 chunks of 32 value-rows (V staged transposed) ----
  v8f O[16];
#pragma unroll
  for (int i = 0; i < 16; ++i) O[i] = (v8f)0.0f;

  for (int tc = 0; tc < 8; ++tc) {
#pragma unroll
    for (int i = 0; i < 4; ++i) {
      int idx = tid + i * 256;
      int t = idx >> 5, dj = (idx & 31) * 8;
      v8h vv = *(const v8h*)(Vg + (size_t)(grp0 + tc * 32 + t) * kD + dj);
#pragma unroll
      for (int e = 0; e < 8; ++e) Vt[(dj + e) * 32 + t] = vv[e];
    }
    __syncthreads();
    v16h pf = load_afrag(Prow, 0, kD, tc * 32, lane);
#pragma unroll
    for (int dt = 0; dt < 16; ++dt) {
      v16h vf = load_bfrag(Vt, dt * 16, 32, 0, lane);
      O[dt] = wmma_f16(pf, vf, O[dt]);
    }
    __syncthreads();
  }

  {
    int n = lane & 15, hh = lane >> 4;
#pragma unroll
    for (int dt = 0; dt < 16; ++dt)
#pragma unroll
      for (int r = 0; r < 8; ++r)
        Og[(size_t)(myrow + r + 8 * hh) * kD + dt * 16 + n] = (_Float16)O[dt][r];
  }
}

// ---------------- kernel 5: O2 [B*N, d] f16 -> out [B,d,N] f32 ------------------------------
__global__ void out_t_kernel(const _Float16* __restrict__ in, float* __restrict__ out) {
  __shared__ float tile[32][33];
  int n0 = blockIdx.x * 32, d0 = blockIdx.y * 32, b = blockIdx.z;
  int tx = threadIdx.x, ty = threadIdx.y;
#pragma unroll
  for (int i = 0; i < 4; ++i) {
    int nn = n0 + ty + i * 8;
    tile[ty + i * 8][tx] = (float)in[(size_t)(b * kN + nn) * kD + d0 + tx];
  }
  __syncthreads();
#pragma unroll
  for (int i = 0; i < 4; ++i) {
    int dd = ty + i * 8;
    out[(size_t)(b * kD + d0 + dd) * kN + n0 + tx] = tile[tx][dd];
  }
}

// ---------------- host launch ---------------------------------------------------------------
extern "C" void kernel_launch(void* const* d_in, const int* in_sizes, int n_in,
                              void* d_out, int out_size, void* d_ws, size_t ws_size,
                              hipStream_t stream) {
  const float* feature = (const float*)d_in[0];
  const float* W       = (const float*)d_in[1];
  const float* b       = (const float*)d_in[2];
  const float* gamma   = (const float*)d_in[3];
  const float* beta    = (const float*)d_in[4];
  const float* mean    = (const float*)d_in[5];
  const float* var     = (const float*)d_in[6];
  // d_in[7] = num_slice (256, hardcoded)

  char* ws = (char*)d_ws;
  _Float16* Wp = (_Float16*)ws;  ws += (size_t)6 * kD * kD * sizeof(_Float16);
  float*    bp = (float*)ws;     ws += (size_t)6 * kD * sizeof(float);
  ws = (char*)(((uintptr_t)ws + 255) & ~(uintptr_t)255);
  const size_t big = (size_t)kPOS * kD * sizeof(_Float16);   // 64 MiB each
  _Float16* Xt = (_Float16*)ws;  ws += big;                  // stage-1 input / stage-2 output
  _Float16* Q  = (_Float16*)ws;  ws += big;
  _Float16* K  = (_Float16*)ws;  ws += big;
  _Float16* V  = (_Float16*)ws;  ws += big;
  _Float16* O1 = (_Float16*)ws;  ws += big;

  // 1) fold BN into weights
  prep_kernel<<<dim3(6 * kD), dim3(kD), 0, stream>>>(W, b, gamma, beta, mean, var, Wp, bp);
  // 2) transpose feature -> [pos, d] f16
  feat_t_kernel<<<dim3(kN / 32, kD / 32, kB), dim3(32, 8), 0, stream>>>(feature, Xt);
  // 3) stage-1 projections (q,k,v), positions ordered (m,s)
  proj_kernel<<<dim3(kPOS / 128, 3), dim3(256), 0, stream>>>(Xt, Wp, bp, Q, K, V, 0);
  // 4) stage-1 attention over s within each (b,m); output O1 ordered (m,s)
  attn_kernel<<<dim3(kB * 256, 2), dim3(256), 0, stream>>>(Q, K, V, O1);
  // 5) stage-2 projections, output positions transposed to (s,m) ordering
  proj_kernel<<<dim3(kPOS / 128, 3), dim3(256), 0, stream>>>(
      O1, Wp + (size_t)3 * kD * kD, bp + 3 * kD, Q, K, V, 1);
  // 6) stage-2 attention over m within each (b,s); output ordered (s,m) = final n
  attn_kernel<<<dim3(kB * 256, 2), dim3(256), 0, stream>>>(Q, K, V, Xt);
  // 7) transpose to f32 [B, d, N, 1]
  out_t_kernel<<<dim3(kN / 32, kD / 32, kB), dim3(32, 8), 0, stream>>>(Xt, (float*)d_out);
}
